// GroupQueryAttention_57758720197303
// MI455X (gfx1250) — matrix operand order repaired and checked by reading, not compiled, not run on hardware
//
#include <hip/hip_runtime.h>
#include <cstdint>

typedef __attribute__((ext_vector_type(16))) _Float16 v16h;
typedef __attribute__((ext_vector_type(8)))  _Float16 v8h;
typedef __attribute__((ext_vector_type(8)))  float    v8f;

constexpr int CB = 2, CS = 2048, CD = 2048, CH = 16, CG = 4, CDH = 128;
constexpr float SM_SCALE = 0.08838834764831845f;        // 1/sqrt(128)
constexpr float LOG_ROPE = 9.210340371976184f;          // ln(10000)

// ---------- CDNA5 async global->LDS staging (probe-guarded) ----------
#if __has_builtin(__builtin_amdgcn_global_load_async_to_lds_b128)
#define ASYNC_LDS 1
typedef __attribute__((__vector_size__(4 * sizeof(int)))) int v4i_t;
typedef __attribute__((address_space(1))) v4i_t GV4;   // global (AS1)
typedef __attribute__((address_space(3))) v4i_t LV4;   // LDS (AS3)
__device__ __forceinline__ void async_lds_b128(const void* gsrc, void* ldst) {
  // integer casts: legal across address spaces; LDS offset = low 32 bits of
  // the generic LDS pointer
  GV4* g = (GV4*)(uintptr_t)gsrc;
  LV4* l = (LV4*)(uint32_t)(uintptr_t)ldst;
  __builtin_amdgcn_global_load_async_to_lds_b128(g, l, 0, 0);
}
#if __has_builtin(__builtin_amdgcn_s_wait_asynccnt)
__device__ __forceinline__ void wait_async0() { __builtin_amdgcn_s_wait_asynccnt(0); }
#else
__device__ __forceinline__ void wait_async0() {
  asm volatile("s_wait_asynccnt 0" ::: "memory");
}
#endif
#endif

// ---------- WMMA fragment helpers (CDNA5 wave32 layouts) ----------
// A 16x32 f16: lane m=lane&15, elements j<8 -> K=kb+j, j>=8 -> K=kb+8+j (kb=(lane>>4)*8)
__device__ __forceinline__ v16h frag_a(const _Float16* p) {
  v8h lo = *(const v8h*)p;
  v8h hi = *(const v8h*)(p + 16);
  v16h r;
#pragma unroll
  for (int j = 0; j < 8; ++j) { r[j] = lo[j]; r[j + 8] = hi[j]; }
  return r;
}
// B 32x16 f16 read from an N-major buffer: same K order as frag_a (p=kb, kb=(lane>>4)*8)
__device__ __forceinline__ v16h frag_b(const _Float16* p) {
  v8h lo = *(const v8h*)p;
  v8h hi = *(const v8h*)(p + 16);
  v16h r;
#pragma unroll
  for (int j = 0; j < 8; ++j) { r[j] = lo[j]; r[j + 8] = hi[j]; }
  return r;
}
__device__ __forceinline__ v8f wmma32(v16h a, v16h b, v8f c) {
  return __builtin_amdgcn_wmma_f32_16x16x32_f16(false, a, false, b, (short)0, c,
                                                false, false);
}

// ---------- f32 -> f16 cast ----------
__global__ void cast_f32_f16(const float* __restrict__ in, _Float16* __restrict__ out,
                             size_t n) {
  size_t i = (size_t)blockIdx.x * blockDim.x + threadIdx.x;
  size_t stride = (size_t)gridDim.x * blockDim.x;
  for (; i < n; i += stride) out[i] = (_Float16)in[i];
}

// ---------- tiled WMMA GEMM: C[M,N] = A[M,K] * B[K,N] ----------
constexpr int BM = 128, BN = 128, BK = 32, BKP = 40;  // pad keeps 16B alignment

template <typename OutT>
__global__ __launch_bounds__(256) void gemm_f16(const _Float16* __restrict__ A,
                                                const _Float16* __restrict__ Bm,
                                                OutT* __restrict__ C,
                                                int M, int N, int K) {
  __shared__ __align__(16) _Float16 As[BM][BKP];
  __shared__ __align__(16) _Float16 Bt[BN][BKP];   // B transposed (N-major)
  const int tid = threadIdx.x;
  const int lane = tid & 31;
  const int wave = tid >> 5;           // 8 waves: 2(M) x 4(N)
  const int wm = (wave >> 2) * 64;
  const int wn = (wave & 3) * 32;
  const int cm0 = blockIdx.y * BM;
  const int cn0 = blockIdx.x * BN;
  const int ml = lane & 15;
  const int hi = lane >> 4;

  const int ra = tid >> 1, ca = (tid & 1) * 16;      // A-tile stage coords
  const int rb = tid >> 3, cb = (tid & 7) * 16;      // B-tile stage coords

  v8f acc[4][2] = {};

  for (int k0 = 0; k0 < K; k0 += BK) {
    const _Float16* srcA = A + (size_t)(cm0 + ra) * K + k0 + ca;
#if defined(ASYNC_LDS)
    // CDNA5 async DMA global->LDS: no VGPR staging, tracked on ASYNCcnt
    async_lds_b128(srcA, &As[ra][ca]);
    async_lds_b128(srcA + 8, &As[ra][ca + 8]);
#else
    *(v8h*)&As[ra][ca] = *(const v8h*)srcA;
    *(v8h*)&As[ra][ca + 8] = *(const v8h*)(srcA + 8);
#endif
    {   // B tile: 32 rows x 128 cols, stored transposed into LDS
      const _Float16* srcB = Bm + (size_t)(k0 + rb) * N + cn0 + cb;
      v8h b0 = *(const v8h*)srcB;
      v8h b1 = *(const v8h*)(srcB + 8);
#pragma unroll
      for (int j = 0; j < 8; ++j) { Bt[cb + j][rb] = b0[j]; Bt[cb + 8 + j][rb] = b1[j]; }
    }
    if (k0 + BK < K) {   // pull next tiles toward L2/WGP$ (global_prefetch_b8)
      __builtin_prefetch(srcA + BK, 0, 1);
      __builtin_prefetch(Bm + (size_t)(k0 + BK + rb) * N + cn0 + cb, 0, 1);
    }
#if defined(ASYNC_LDS)
    wait_async0();
#endif
    __syncthreads();
    v16h af[4], bf[2];
#pragma unroll
    for (int i = 0; i < 4; ++i) af[i] = frag_a(&As[wm + i * 16 + ml][hi * 8]);
#pragma unroll
    for (int j = 0; j < 2; ++j) bf[j] = frag_b(&Bt[wn + j * 16 + ml][hi * 8]);
#pragma unroll
    for (int i = 0; i < 4; ++i)
#pragma unroll
      for (int j = 0; j < 2; ++j) acc[i][j] = wmma32(af[i], bf[j], acc[i][j]);
    __syncthreads();
  }
#pragma unroll
  for (int i = 0; i < 4; ++i)
#pragma unroll
    for (int j = 0; j < 2; ++j)
#pragma unroll
      for (int r = 0; r < 8; ++r)
        C[(size_t)(cm0 + wm + i * 16 + hi * 8 + r) * N + cn0 + wn + j * 16 + ml] =
            (OutT)acc[i][j][r];
}

// ---------- RoPE + layout transforms ----------
// Qp f16 [B,S,H,DH] -> Qh f16 [B,H,S,DH] (roped)
__global__ void rope_q_kernel(const _Float16* __restrict__ Qp,
                              _Float16* __restrict__ Qh) {
  int idx = blockIdx.x * blockDim.x + threadIdx.x;
  if (idx >= CB * CH * CS * (CDH / 2)) return;
  int i = idx % 64;
  int s = (idx / 64) % CS;
  int h = (idx / (64 * CS)) % CH;
  int b = idx / (64 * CS * CH);
  float ang = (float)s * __expf(-(float)i * (LOG_ROPE / 64.0f));
  float sn, cs;
  __sincosf(ang, &sn, &cs);
  size_t in0 = (((size_t)b * CS + s) * CH + h) * CDH + i;
  float x1 = (float)Qp[in0], x2 = (float)Qp[in0 + 64];
  size_t o0 = (((size_t)b * CH + h) * CS + s) * CDH + i;
  Qh[o0] = (_Float16)(x1 * cs - x2 * sn);
  Qh[o0 + 64] = (_Float16)(x2 * cs + x1 * sn);
}

// Kp f16 [B,S,G,DH] -> Kh f16 [B,G,S,DH] (roped) + f32 k-cache [B,G,S,DH]
__global__ void rope_k_kernel(const _Float16* __restrict__ Kp,
                              _Float16* __restrict__ Kh, float* __restrict__ kc) {
  int idx = blockIdx.x * blockDim.x + threadIdx.x;
  if (idx >= CB * CG * CS * (CDH / 2)) return;
  int i = idx % 64;
  int s = (idx / 64) % CS;
  int g = (idx / (64 * CS)) % CG;
  int b = idx / (64 * CS * CG);
  float ang = (float)s * __expf(-(float)i * (LOG_ROPE / 64.0f));
  float sn, cs;
  __sincosf(ang, &sn, &cs);
  size_t in0 = (((size_t)b * CS + s) * CG + g) * CDH + i;
  float x1 = (float)Kp[in0], x2 = (float)Kp[in0 + 64];
  float r1 = x1 * cs - x2 * sn, r2 = x2 * cs + x1 * sn;
  size_t o0 = (((size_t)b * CG + g) * CS + s) * CDH + i;
  Kh[o0] = (_Float16)r1;
  Kh[o0 + 64] = (_Float16)r2;
  kc[o0] = r1;
  kc[o0 + 64] = r2;
}

// Vp f16 [B,S,G,DH] -> VT f16 [B,G,DH,S] + f32 v-cache [B,G,S,DH]
__global__ void v_prep_kernel(const _Float16* __restrict__ Vp,
                              _Float16* __restrict__ VT, float* __restrict__ vc) {
  int idx = blockIdx.x * blockDim.x + threadIdx.x;
  if (idx >= CB * CG * CS * CDH) return;
  int d = idx % CDH;
  int s = (idx / CDH) % CS;
  int g = (idx / (CDH * CS)) % CG;
  int b = idx / (CDH * CS * CG);
  float v = (float)Vp[(((size_t)b * CS + s) * CG + g) * CDH + d];
  vc[(((size_t)b * CG + g) * CS + s) * CDH + d] = v;
  VT[(((size_t)b * CG + g) * CDH + d) * CS + s] = (_Float16)v;
}

// ---------- flash attention: one wave per 16-query tile ----------
// Computes S^T = K*Q^T so the softmax reduction (over keys) lives in per-lane
// VGPRs + one lane^16 shuffle. P is staged via LDS into the A-fragment layout.
// All K and V fragments for a 32-key chunk are preloaded into distinct
// registers so global-load latency overlaps the WMMA + softmax work.
__global__ __launch_bounds__(32) void attn_fwd(const _Float16* __restrict__ Qh,
                                               const _Float16* __restrict__ Kh,
                                               const _Float16* __restrict__ VT,
                                               _Float16* __restrict__ ctx) {
  __shared__ __align__(16) _Float16 P[16][40];
  const int lane = threadIdx.x;
  const int ql = lane & 15;
  const int hi = lane >> 4;
  const int nq = CS / 16;
  const int qt = blockIdx.x % nq;
  const int h = (blockIdx.x / nq) % CH;
  const int b = blockIdx.x / (nq * CH);
  const int g = h / (CH / CG);
  const int q0 = qt * 16;

  const _Float16* Qb = Qh + (((size_t)b * CH + h) * CS) * CDH;
  const _Float16* Kb = Kh + (((size_t)b * CG + g) * CS) * CDH;
  const _Float16* Vb = VT + (((size_t)b * CG + g) * CDH) * CS;

  v16h qb[4];   // Q^T as B-fragments (lane = query column, contiguous d)
#pragma unroll
  for (int dk = 0; dk < 4; ++dk)
    qb[dk] = frag_b(Qb + (size_t)(q0 + ql) * CDH + dk * 32 + hi * 8);

  v8f acc[8] = {};                 // ctx accumulators, rows = queries
  float mrow = -1e30f, lrow = 0.f; // per-query stats (query = ql in both halves)

  for (int kc = 0; kc < q0 + 16; kc += 32) {
    // ---- preload all fragments for this chunk (loads issue as clauses) ----
    v16h kf[2][4];
#pragma unroll
    for (int t = 0; t < 2; ++t)
#pragma unroll
      for (int dk = 0; dk < 4; ++dk)
        kf[t][dk] = frag_a(Kb + (size_t)(kc + t * 16 + ql) * CDH + dk * 32 + hi * 8);
    v16h vf[8];   // V fragments are independent of scores: latency hidden
#pragma unroll
    for (int n = 0; n < 8; ++n)
      vf[n] = frag_b(Vb + (size_t)(n * 16 + ql) * CS + kc + hi * 8);
    if (kc + 32 < q0 + 16)   // prefetch next key chunk toward L2
      __builtin_prefetch(Kb + (size_t)(kc + 32 + ql) * CDH, 0, 1);

    // ---- scores: S^T tile (row=key, col=query) ----
    float sreg[2][8];
#pragma unroll
    for (int t = 0; t < 2; ++t) {
      v8f sc = {};
#pragma unroll
      for (int dk = 0; dk < 4; ++dk) sc = wmma32(kf[t][dk], qb[dk], sc);
#pragma unroll
      for (int r = 0; r < 8; ++r) {
        int kk = kc + t * 16 + hi * 8 + r;           // global key index
        float v = sc[r] * SM_SCALE;
        sreg[t][r] = (kk > q0 + ql) ? -1e30f : v;    // causal mask
      }
    }
    // ---- online softmax over keys ----
    float cmax = -1e30f;
#pragma unroll
    for (int t = 0; t < 2; ++t)
#pragma unroll
      for (int r = 0; r < 8; ++r) cmax = fmaxf(cmax, sreg[t][r]);
    cmax = fmaxf(cmax, __shfl_xor(cmax, 16, 32));
    float mnew = fmaxf(mrow, cmax);
    float alpha = __expf(mrow - mnew);
    float csum = 0.f;
#pragma unroll
    for (int t = 0; t < 2; ++t)
#pragma unroll
      for (int r = 0; r < 8; ++r) {
        float pv = __expf(sreg[t][r] - mnew);
        sreg[t][r] = pv;
        csum += pv;
      }
    csum += __shfl_xor(csum, 16, 32);
    lrow = lrow * alpha + csum;
    mrow = mnew;

    // rescale ctx accumulators: row M = r + hi*8 needs alpha of that query
    float av[8];
#pragma unroll
    for (int r = 0; r < 8; ++r) av[r] = __shfl(alpha, r + hi * 8, 32);
#pragma unroll
    for (int n = 0; n < 8; ++n)
#pragma unroll
      for (int r = 0; r < 8; ++r) acc[n][r] *= av[r];

    // stage P (16q x 32k) through LDS to re-enter the A-fragment layout
    __syncthreads();
#pragma unroll
    for (int t = 0; t < 2; ++t)
#pragma unroll
      for (int r = 0; r < 8; ++r)
        P[ql][t * 16 + hi * 8 + r] = (_Float16)sreg[t][r];
    __syncthreads();

    v16h pf = frag_a(&P[ql][hi * 8]);
#pragma unroll
    for (int n = 0; n < 8; ++n)   // ctx += P * V over DH = 8 n-tiles
      acc[n] = wmma32(pf, vf[n], acc[n]);
  }

  float lv[8];
#pragma unroll
  for (int r = 0; r < 8; ++r) lv[r] = 1.0f / __shfl(lrow, r + hi * 8, 32);
#pragma unroll
  for (int n = 0; n < 8; ++n)
#pragma unroll
    for (int r = 0; r < 8; ++r)
      ctx[(((size_t)b * CS + q0 + hi * 8 + r) * CH + h) * CDH + n * 16 + ql] =
          (_Float16)(acc[n][r] * lv[r]);
}

// ---------- host ----------
extern "C" void kernel_launch(void* const* d_in, const int* in_sizes, int n_in,
                              void* d_out, int out_size, void* d_ws, size_t ws_size,
                              hipStream_t stream) {
  (void)in_sizes; (void)n_in; (void)out_size; (void)ws_size;
  const float* x  = (const float*)d_in[0];
  const float* Wq = (const float*)d_in[1];
  const float* Wk = (const float*)d_in[2];
  const float* Wv = (const float*)d_in[3];
  const float* Wo = (const float*)d_in[4];
  // d_in[5] = mask (causal triu) — applied analytically in attn_fwd

  float* out = (float*)d_out;                                 // [B,S,D]
  float* kcache = out + (size_t)CB * CS * CD;                 // [B,G,S,DH]
  float* vcache = kcache + (size_t)CB * CG * CS * CDH;        // [B,G,S,DH]

  char* p = (char*)d_ws;
  auto take = [&](size_t bytes) {
    char* r = p;
    p += (bytes + 255) & ~(size_t)255;
    return r;
  };
  const size_t n_x  = (size_t)CB * CS * CD;
  const size_t n_wq = (size_t)CD * CH * CDH;
  const size_t n_wk = (size_t)CD * CG * CDH;
  const size_t n_q  = (size_t)CB * CS * CH * CDH;
  const size_t n_k  = (size_t)CB * CS * CG * CDH;

  _Float16* xh  = (_Float16*)take(n_x * 2);
  _Float16* Wqh = (_Float16*)take(n_wq * 2);
  _Float16* Wkh = (_Float16*)take(n_wk * 2);
  _Float16* Wvh = (_Float16*)take(n_wk * 2);
  _Float16* Woh = (_Float16*)take(n_wq * 2);
  _Float16* Qp  = (_Float16*)take(n_q * 2);   // [B,S,H,DH] pre-rope
  _Float16* Kp  = (_Float16*)take(n_k * 2);   // [B,S,G,DH] pre-rope
  _Float16* Vp  = (_Float16*)take(n_k * 2);   // [B,S,G,DH]
  _Float16* Qh  = (_Float16*)take(n_q * 2);   // [B,H,S,DH] roped
  _Float16* Kh  = (_Float16*)take(n_k * 2);   // [B,G,S,DH] roped
  _Float16* VTh = (_Float16*)take(n_k * 2);   // [B,G,DH,S]
  _Float16* ctxh = Qp;                        // alias: Qp dead after rope_q

  // 1) cast inputs to f16
  cast_f32_f16<<<8192, 256, 0, stream>>>(x, xh, n_x);
  cast_f32_f16<<<4096, 256, 0, stream>>>(Wq, Wqh, n_wq);
  cast_f32_f16<<<2048, 256, 0, stream>>>(Wk, Wkh, n_wk);
  cast_f32_f16<<<2048, 256, 0, stream>>>(Wv, Wvh, n_wk);
  cast_f32_f16<<<4096, 256, 0, stream>>>(Wo, Woh, n_wq);

  // 2) projections (M = B*S = 4096)
  gemm_f16<_Float16><<<dim3(CD / BN, (CB * CS) / BM), 256, 0, stream>>>(
      xh, Wqh, Qp, CB * CS, CH * CDH, CD);
  gemm_f16<_Float16><<<dim3((CG * CDH) / BN, (CB * CS) / BM), 256, 0, stream>>>(
      xh, Wkh, Kp, CB * CS, CG * CDH, CD);
  gemm_f16<_Float16><<<dim3((CG * CDH) / BN, (CB * CS) / BM), 256, 0, stream>>>(
      xh, Wvh, Vp, CB * CS, CG * CDH, CD);

  // 3) RoPE + transposes + f32 caches
  rope_q_kernel<<<(CB * CH * CS * 64) / 256, 256, 0, stream>>>(Qp, Qh);
  rope_k_kernel<<<(CB * CG * CS * 64) / 256, 256, 0, stream>>>(Kp, Kh, kcache);
  v_prep_kernel<<<(CB * CG * CS * CDH) / 256, 256, 0, stream>>>(Vp, VTh, vcache);

  // 4) flash attention: one wave per (b, h, 16-query tile)
  attn_fwd<<<CB * CH * (CS / 16), 32, 0, stream>>>(Qh, Kh, VTh, ctxh);

  // 5) output projection
  gemm_f16<float><<<dim3(CD / BN, (CB * CS) / BM), 256, 0, stream>>>(
      ctxh, Woh, out, CB * CS, CD, CH * CDH);
}
